// PACE_VAE_nomask_58205396795634
// MI455X (gfx1250) — compile-verified, matrix operand order checked by reading, not run on hardware
//
#include <hip/hip_runtime.h>
#include <hip/hip_bf16.h>

// Problem constants (from reference): L=1024, B=8, E=1024, H=16, head_dim=64
#define L_  1024
#define B_  8
#define E_  1024
#define H_  16
#define HD_ 64
#define M_  (L_ * B_)   // 8192 rows (l*B+b)
#define N3_ (3 * E_)    // 3072
#define G_  (B_ * H_)   // 128 heads total

typedef __bf16 bf16_t;
typedef bf16_t v8bf  __attribute__((ext_vector_type(8)));
typedef bf16_t v16bf __attribute__((ext_vector_type(16)));
typedef float  v8f   __attribute__((ext_vector_type(8)));

// A-fragment (16x32 MxK bf16) per-lane layout (ISA 7.12.2):
//   lane l (row m=l&15, half=l>>4) holds K = half*8 + {0..7} and 16 + half*8 + {0..7}
// B-fragment (32x16 KxN) is the symmetric per-lane column layout.
// Both reduce to two contiguous 16-byte loads per lane.
static __device__ __forceinline__ v16bf frag16(const bf16_t* p0, const bf16_t* p1) {
  v8bf a = *reinterpret_cast<const v8bf*>(p0);
  v8bf b = *reinterpret_cast<const v8bf*>(p1);
  v16bf r;
#pragma unroll
  for (int i = 0; i < 8; ++i) { r[i] = a[i]; r[i + 8] = b[i]; }
  return r;
}

#define WMMA_BF16(Afrag, Bfrag, Cfrag) \
  __builtin_amdgcn_wmma_f32_16x16x32_bf16(false, (Afrag), false, (Bfrag), (short)0, (Cfrag), false, false)

// ---------------------------------------------------------------- conversions
__global__ void cvt_bf16_kernel(const float* __restrict__ s, bf16_t* __restrict__ d, int n) {
  int i = blockIdx.x * blockDim.x + threadIdx.x;
  if (i < n) d[i] = (bf16_t)s[i];
}

// ---------------------------------------------------------------- QKV GEMM
// C[m,n] = sum_k X[m,k] * W[n,k] + bias[n]; M=8192, N=3072, K=1024.
// Epilogue scatters into per-head Q/K ([g][l][d]) and transposed V ([g][d][l]).
__global__ void __launch_bounds__(256) qkv_gemm_kernel(
    const bf16_t* __restrict__ X, const bf16_t* __restrict__ W,
    const float* __restrict__ bias,
    bf16_t* __restrict__ Qb, bf16_t* __restrict__ Kb, bf16_t* __restrict__ Vt) {
  const int wave = (int)((blockIdx.x * blockDim.x + threadIdx.x) >> 5);
  const int lane = threadIdx.x & 31;
  const int l16 = lane & 15, half = lane >> 4;
  const int NT = N3_ / 64;            // 48 n-tiles
  const int mt = wave / NT, nt = wave % NT;
  if (mt >= M_ / 64) return;
  const int m0 = mt * 64, n0 = nt * 64;

  v8f acc[4][4] = {};

  for (int k0 = 0; k0 < E_; k0 += 32) {
    if (k0 + 32 < E_) {
      __builtin_prefetch(X + (size_t)(m0 + l16) * E_ + k0 + 32, 0, 3);
      __builtin_prefetch(W + (size_t)(n0 + l16) * E_ + k0 + 32, 0, 3);
    }
    v16bf a[4], b[4];
#pragma unroll
    for (int i = 0; i < 4; ++i) {
      const bf16_t* r = X + (size_t)(m0 + i * 16 + l16) * E_ + k0 + half * 8;
      a[i] = frag16(r, r + 16);
    }
#pragma unroll
    for (int j = 0; j < 4; ++j) {
      const bf16_t* r = W + (size_t)(n0 + j * 16 + l16) * E_ + k0 + half * 8;
      b[j] = frag16(r, r + 16);
    }
#pragma unroll
    for (int i = 0; i < 4; ++i)
#pragma unroll
      for (int j = 0; j < 4; ++j)
        acc[i][j] = WMMA_BF16(a[i], b[j], acc[i][j]);
  }

  // n-tile of 64 cols lies fully inside one of {Q,K,V} and one head (hd=64).
  const int sect = n0 >> 10;            // 0=Q, 1=K, 2=V
  const int h    = (n0 & 1023) >> 6;    // head-within-embed
#pragma unroll
  for (int j = 0; j < 4; ++j) {
    const int n  = n0 + j * 16 + l16;
    const float bv = bias[n];
    const int d  = j * 16 + l16;        // 0..63 within head
#pragma unroll
    for (int i = 0; i < 4; ++i) {
#pragma unroll
      for (int r = 0; r < 8; ++r) {
        const int m  = m0 + i * 16 + r + 8 * half;   // m = l*B + b
        const int lq = m >> 3;
        const int bb = m & 7;
        const int g  = bb * H_ + h;
        const float c = acc[i][j][r] + bv;
        if (sect == 0)      Qb[((size_t)g * L_ + lq) * HD_ + d] = (bf16_t)(c * 0.125f);
        else if (sect == 1) Kb[((size_t)g * L_ + lq) * HD_ + d] = (bf16_t)c;
        else                Vt[((size_t)g * HD_ + d) * L_ + lq] = (bf16_t)c;
      }
    }
  }
}

// ---------------------------------------------------------------- attention
// One wave per (head g, 16-query tile qt). Flash-style online softmax over
// 32-key steps: 4 WMMAs for S, shuffle reductions, P via LDS relayout, 4 WMMAs
// for O += P*V. Writes bf16 context in [m=l*B+b][E] layout.
__global__ void __launch_bounds__(256) attn_kernel(
    const bf16_t* __restrict__ Qb, const bf16_t* __restrict__ Kb,
    const bf16_t* __restrict__ Vt, bf16_t* __restrict__ Ctx) {
  __shared__ __align__(16) bf16_t plds[8 * 16 * 32];  // 1KB per wave
  const int lane = threadIdx.x & 31;
  const int wid  = threadIdx.x >> 5;
  const int wave = blockIdx.x * 8 + wid;
  const int g  = wave >> 6;   // head 0..127
  const int qt = wave & 63;   // query tile 0..63
  if (g >= G_) return;
  const int l16 = lane & 15, half = lane >> 4;

  const bf16_t* Qh = Qb + (size_t)g * L_ * HD_;
  const bf16_t* Kh = Kb + (size_t)g * L_ * HD_;
  const bf16_t* Vh = Vt + (size_t)g * HD_ * L_;
  bf16_t* myP = plds + wid * (16 * 32);

  v16bf qf[2];
#pragma unroll
  for (int kc = 0; kc < 2; ++kc) {
    const bf16_t* r = Qh + (size_t)(qt * 16 + l16) * HD_ + kc * 32 + half * 8;
    qf[kc] = frag16(r, r + 16);
  }

  v8f o0 = {}, o1 = {}, o2 = {}, o3 = {};
  float mrun[8], lrun[8];
#pragma unroll
  for (int r = 0; r < 8; ++r) { mrun[r] = -3.0e38f; lrun[r] = 0.0f; }

  for (int kb = 0; kb < L_; kb += 32) {
    // S = Q * K^T for 32 keys -> two 16x16 tiles
    v8f s0 = {}, s1 = {};
#pragma unroll
    for (int kc = 0; kc < 2; ++kc) {
      const bf16_t* r0 = Kh + (size_t)(kb + l16) * HD_ + kc * 32 + half * 8;
      const bf16_t* r1 = Kh + (size_t)(kb + 16 + l16) * HD_ + kc * 32 + half * 8;
      v16bf kf0 = frag16(r0, r0 + 16);
      v16bf kf1 = frag16(r1, r1 + 16);
      s0 = WMMA_BF16(qf[kc], kf0, s0);
      s1 = WMMA_BF16(qf[kc], kf1, s1);
    }
    // online softmax: rows live across 16-lane halves (C layout m = r + 8*half)
    float alpha[8];
#pragma unroll
    for (int r = 0; r < 8; ++r) {
      float v = fmaxf(s0[r], s1[r]);
      v = fmaxf(v, __shfl_xor(v, 1));
      v = fmaxf(v, __shfl_xor(v, 2));
      v = fmaxf(v, __shfl_xor(v, 4));
      v = fmaxf(v, __shfl_xor(v, 8));
      const float mnew = fmaxf(mrun[r], v);
      alpha[r] = __expf(mrun[r] - mnew);
      const float p0 = __expf(s0[r] - mnew);
      const float p1 = __expf(s1[r] - mnew);
      s0[r] = p0; s1[r] = p1;
      float rs = p0 + p1;
      rs += __shfl_xor(rs, 1);
      rs += __shfl_xor(rs, 2);
      rs += __shfl_xor(rs, 4);
      rs += __shfl_xor(rs, 8);
      lrun[r] = lrun[r] * alpha[r] + rs;
      mrun[r] = mnew;
    }
#pragma unroll
    for (int r = 0; r < 8; ++r) {
      o0[r] *= alpha[r]; o1[r] *= alpha[r]; o2[r] *= alpha[r]; o3[r] *= alpha[r];
    }
    // P (C layout, f32) -> LDS row-major 16x32 bf16 -> A-fragment
#pragma unroll
    for (int r = 0; r < 8; ++r) {
      myP[(r + 8 * half) * 32 + l16]      = (bf16_t)s0[r];
      myP[(r + 8 * half) * 32 + 16 + l16] = (bf16_t)s1[r];
    }
    asm volatile("s_wait_dscnt 0x0" ::: "memory");  // DS in-order per wave; fence compiler too
    const bf16_t* pr = myP + l16 * 32 + half * 8;
    v16bf pf = frag16(pr, pr + 16);
    // O += P * V  (V transposed: [d][l], so B columns are contiguous along keys)
#pragma unroll
    for (int t = 0; t < 4; ++t) {
      const bf16_t* vr = Vh + (size_t)(t * 16 + l16) * L_ + kb + half * 8;
      v16bf vf = frag16(vr, vr + 16);
      if      (t == 0) o0 = WMMA_BF16(pf, vf, o0);
      else if (t == 1) o1 = WMMA_BF16(pf, vf, o1);
      else if (t == 2) o2 = WMMA_BF16(pf, vf, o2);
      else             o3 = WMMA_BF16(pf, vf, o3);
    }
  }

  // normalize and store context in [m = l*B + b][E] bf16 layout
  const int bb = g >> 4, h = g & 15;
  float inv[8];
#pragma unroll
  for (int r = 0; r < 8; ++r) inv[r] = 1.0f / lrun[r];
#pragma unroll
  for (int r = 0; r < 8; ++r) {
    const int mq = qt * 16 + r + 8 * half;
    const size_t rowbase = ((size_t)mq * B_ + bb) * E_ + h * 64;
    Ctx[rowbase +  0 + l16] = (bf16_t)(o0[r] * inv[r]);
    Ctx[rowbase + 16 + l16] = (bf16_t)(o1[r] * inv[r]);
    Ctx[rowbase + 32 + l16] = (bf16_t)(o2[r] * inv[r]);
    Ctx[rowbase + 48 + l16] = (bf16_t)(o3[r] * inv[r]);
  }
}

// ---------------------------------------------------------------- out proj
// out[m,n] = sum_k Ctx[m,k] * Wout[n,k] + bias[n]; f32 output = d_out layout.
__global__ void __launch_bounds__(256) out_gemm_kernel(
    const bf16_t* __restrict__ A, const bf16_t* __restrict__ W,
    const float* __restrict__ bias, float* __restrict__ out) {
  const int wave = (int)((blockIdx.x * blockDim.x + threadIdx.x) >> 5);
  const int lane = threadIdx.x & 31;
  const int l16 = lane & 15, half = lane >> 4;
  const int NT = E_ / 64;             // 16 n-tiles
  const int mt = wave / NT, nt = wave % NT;
  if (mt >= M_ / 64) return;
  const int m0 = mt * 64, n0 = nt * 64;

  v8f acc[4][4] = {};

  for (int k0 = 0; k0 < E_; k0 += 32) {
    if (k0 + 32 < E_) {
      __builtin_prefetch(A + (size_t)(m0 + l16) * E_ + k0 + 32, 0, 3);
      __builtin_prefetch(W + (size_t)(n0 + l16) * E_ + k0 + 32, 0, 3);
    }
    v16bf a[4], b[4];
#pragma unroll
    for (int i = 0; i < 4; ++i) {
      const bf16_t* r = A + (size_t)(m0 + i * 16 + l16) * E_ + k0 + half * 8;
      a[i] = frag16(r, r + 16);
    }
#pragma unroll
    for (int j = 0; j < 4; ++j) {
      const bf16_t* r = W + (size_t)(n0 + j * 16 + l16) * E_ + k0 + half * 8;
      b[j] = frag16(r, r + 16);
    }
#pragma unroll
    for (int i = 0; i < 4; ++i)
#pragma unroll
      for (int j = 0; j < 4; ++j)
        acc[i][j] = WMMA_BF16(a[i], b[j], acc[i][j]);
  }

#pragma unroll
  for (int j = 0; j < 4; ++j) {
    const int n = n0 + j * 16 + l16;
    const float bv = bias[n];
#pragma unroll
    for (int i = 0; i < 4; ++i)
#pragma unroll
      for (int r = 0; r < 8; ++r)
        out[(size_t)(m0 + i * 16 + r + 8 * half) * E_ + n] = acc[i][j][r] + bv;
  }
}

// ---------------------------------------------------------------- launch
extern "C" void kernel_launch(void* const* d_in, const int* in_sizes, int n_in,
                              void* d_out, int out_size, void* d_ws, size_t ws_size,
                              hipStream_t stream) {
  const float* x    = (const float*)d_in[0];   // [L,B,E]
  const float* wqkv = (const float*)d_in[1];   // [3E,E]
  const float* bqkv = (const float*)d_in[2];   // [3E]
  const float* wout = (const float*)d_in[3];   // [E,E]
  const float* bout = (const float*)d_in[4];   // [E]
  float* out = (float*)d_out;                  // [L,B,E] f32

  char* ws = (char*)d_ws;
  size_t off = 0;
  bf16_t* Xbf = (bf16_t*)(ws + off); off += (size_t)M_ * E_ * sizeof(bf16_t);
  bf16_t* Wq  = (bf16_t*)(ws + off); off += (size_t)N3_ * E_ * sizeof(bf16_t);
  bf16_t* Wo  = (bf16_t*)(ws + off); off += (size_t)E_ * E_ * sizeof(bf16_t);
  bf16_t* Qb  = (bf16_t*)(ws + off); off += (size_t)G_ * L_ * HD_ * sizeof(bf16_t);
  bf16_t* Kb  = (bf16_t*)(ws + off); off += (size_t)G_ * L_ * HD_ * sizeof(bf16_t);
  bf16_t* Vt  = (bf16_t*)(ws + off); off += (size_t)G_ * HD_ * L_ * sizeof(bf16_t);
  bf16_t* Ctx = (bf16_t*)(ws + off); off += (size_t)M_ * E_ * sizeof(bf16_t);
  (void)ws_size; (void)in_sizes; (void)n_in; (void)out_size;  // ~88MB used

  const int n1 = M_ * E_;
  const int n2 = N3_ * E_;
  const int n3 = E_ * E_;
  cvt_bf16_kernel<<<(n1 + 255) / 256, 256, 0, stream>>>(x, Xbf, n1);
  cvt_bf16_kernel<<<(n2 + 255) / 256, 256, 0, stream>>>(wqkv, Wq, n2);
  cvt_bf16_kernel<<<(n3 + 255) / 256, 256, 0, stream>>>(wout, Wo, n3);

  // QKV GEMM: (8192/64)*(3072/64) = 6144 waves, 8 waves/block
  qkv_gemm_kernel<<<6144 / 8, 256, 0, stream>>>(Xbf, Wq, bqkv, Qb, Kb, Vt);

  // Attention: 128 heads * 64 query tiles = 8192 waves
  attn_kernel<<<8192 / 8, 256, 0, stream>>>(Qb, Kb, Vt, Ctx);

  // Out projection: (8192/64)*(1024/64) = 2048 waves
  out_gemm_kernel<<<2048 / 8, 256, 0, stream>>>(Ctx, Wo, bout, out);
}
